// K_attention_64931315581180
// MI455X (gfx1250) — matrix operand order repaired and checked by reading, not compiled, not run on hardware
//
#include <hip/hip_runtime.h>
#include <hip/hip_bf16.h>
#include <stdint.h>

// CDNA5 (gfx1250) wave32 WMMA types
typedef __attribute__((ext_vector_type(16))) __bf16 v16bf;
typedef __attribute__((ext_vector_type(2)))  __bf16 v2bf;
typedef __attribute__((ext_vector_type(8)))  float  v8f;

union FragBF { v16bf v; uint4 u[2]; };

// f32 -> bf16 round-to-nearest-even (fallback path)
__device__ __forceinline__ unsigned short f2bf(float f) {
  union { float f; unsigned int u; } c; c.f = f;
  unsigned int u = c.u;
  u += 0x7FFFu + ((u >> 16) & 1u);
  return (unsigned short)(u >> 16);
}

// pack two f32 -> packed bf16x2 dword (v_cvt_pk_bf16_f32 when available)
__device__ __forceinline__ unsigned int pk_bf16(float a, float b) {
#if __has_builtin(__builtin_amdgcn_cvt_pk_bf16_f32)
  union { v2bf v; unsigned int u; } c;
  c.v = __builtin_amdgcn_cvt_pk_bf16_f32(a, b);
  return c.u;
#else
  return (unsigned int)f2bf(a) | ((unsigned int)f2bf(b) << 16);
#endif
}

// A-operand fragment (16x32 bf16, ISA 7.12.2): lane holds row (lane&15);
// lanes 0-15: K = kbase+[0..7] in V0-3, kbase+16+[0..7] in V4-7
// lanes 16-31: K = kbase+8+[0..7],      kbase+24+[0..7]
__device__ __forceinline__ v16bf load_frag_A(const unsigned short* t, int row,
                                             int rs, int kbase, int half) {
  FragBF f;
  f.u[0] = *(const uint4*)(t + row * rs + kbase + half * 8);
  f.u[1] = *(const uint4*)(t + row * rs + kbase + 16 + half * 8);
  return f.v;
}

// B-operand fragment (32x16 bf16): lane holds column (lane&15);
// lanes 0-15: K = kbase+[0..7], kbase+16+[0..7]; lanes 16-31: K = kbase+8+[0..7], kbase+24+[0..7]
__device__ __forceinline__ v16bf load_frag_B(const unsigned short* t, int row,
                                             int rs, int kbase, int half) {
  FragBF f;
  f.u[0] = *(const uint4*)(t + row * rs + kbase + half * 8);
  f.u[1] = *(const uint4*)(t + row * rs + kbase + 16 + half * 8);
  return f.v;
}

// out[b,m,c] = x[b,m,c] + sum_s exp(min(2sig*S - sig|x_m|^2 - sig|x_s|^2, 0)) * x[b,s,c]
__global__ __launch_bounds__(256, 1) void gauss_kernel_attn(
    const float* __restrict__ x, const float* __restrict__ r_sigma,
    float* __restrict__ out) {
  constexpr int T = 2048, C = 64;
  constexpr int BM = 128;  // M-rows per block (8 waves x 16)
  constexpr int BS = 64;   // s-chunk
  constexpr int ST = 72;   // LDS row stride in bf16 elems (144B, 16B aligned)

  __shared__ __align__(16) unsigned short xm[BM * ST];     // block's M-rows, bf16
  __shared__ __align__(16) unsigned short xs[BS * ST];     // s-chunk row-major
  __shared__ __align__(16) unsigned short xT[C * ST];      // s-chunk transposed [c][s']
  __shared__ __align__(16) unsigned short kt[8][16 * ST];  // per-wave K tile [m][s']
  __shared__ __align__(16) float sqm[BM];                  // -sigma*|x_m|^2
  __shared__ __align__(16) float sqs[BS];                  // -sigma*|x_s|^2

  const int b    = blockIdx.y;
  const int m0   = blockIdx.x * BM;
  const int tid  = threadIdx.x;
  const int wid  = tid >> 5;
  const int lane = tid & 31;
  const int half = lane >> 4;
  const int ln   = lane & 15;
  const float sigma  = r_sigma[0];
  const float twosig = 2.0f * sigma;
  const float negsig = -sigma;

  const float* xb = x + (size_t)b * T * C;

  // ---- Stage 0: block's 128 M-rows -> LDS bf16, plus -sigma*|row|^2 ----
  if (tid < BM) sqm[tid] = 0.0f;
  __syncthreads();
  {
    const int j  = tid >> 1;        // 0..127
    const int c0 = (tid & 1) * 32;  // 0 or 32
    const float* src = xb + (size_t)(m0 + j) * C + c0;
    float s = 0.0f;
#pragma unroll
    for (int i = 0; i < 32; i += 4) {
      float4 v = *(const float4*)(src + i);
      s += v.x * v.x + v.y * v.y + v.z * v.z + v.w * v.w;
      uint2 p;
      p.x = pk_bf16(v.x, v.y);
      p.y = pk_bf16(v.z, v.w);
      *(uint2*)(&xm[j * ST + c0 + i]) = p;
    }
    atomicAdd(&sqm[j], negsig * s);  // ds_add_f32
  }
  __syncthreads();

  // Loop-invariant per wave: GEMM1 B-operand = this wave's 16 M-rows as columns
  const v16bf bm_lo = load_frag_B(xm, wid * 16 + ln, ST, 0,  half);
  const v16bf bm_hi = load_frag_B(xm, wid * 16 + ln, ST, 32, half);
  const float am    = sqm[wid * 16 + ln];  // lane-constant -sigma*|x_m|^2

  v8f accT[4] = {{}, {}, {}, {}};  // transposed out accumulator: [c-tile][r]

  for (int s0 = 0; s0 < T; s0 += BS) {
    __syncthreads();  // previous iteration's LDS consumers are done
    if (tid < BS) sqs[tid] = 0.0f;
    __syncthreads();

    // ---- stage x[s0:s0+64,:]: 4x4 micro-tiles -> xs (row-major) + xT (transposed)
    {
      const int j0 = (tid >> 4) * 4;  // 0..60
      const int c0 = (tid & 15) * 4;  // 0..60
      const float* src = xb + (size_t)(s0 + j0) * C + c0;
      const float4 r0 = *(const float4*)(src);
      const float4 r1 = *(const float4*)(src + C);
      const float4 r2 = *(const float4*)(src + 2 * C);
      const float4 r3 = *(const float4*)(src + 3 * C);
      // row-major tile (packed b64 stores)
      { uint2 p; p.x = pk_bf16(r0.x, r0.y); p.y = pk_bf16(r0.z, r0.w);
        *(uint2*)(&xs[(j0 + 0) * ST + c0]) = p; }
      { uint2 p; p.x = pk_bf16(r1.x, r1.y); p.y = pk_bf16(r1.z, r1.w);
        *(uint2*)(&xs[(j0 + 1) * ST + c0]) = p; }
      { uint2 p; p.x = pk_bf16(r2.x, r2.y); p.y = pk_bf16(r2.z, r2.w);
        *(uint2*)(&xs[(j0 + 2) * ST + c0]) = p; }
      { uint2 p; p.x = pk_bf16(r3.x, r3.y); p.y = pk_bf16(r3.z, r3.w);
        *(uint2*)(&xs[(j0 + 3) * ST + c0]) = p; }
      // transposed tile (4 consecutive s' per channel row -> packed b64)
      { uint2 p; p.x = pk_bf16(r0.x, r1.x); p.y = pk_bf16(r2.x, r3.x);
        *(uint2*)(&xT[(c0 + 0) * ST + j0]) = p; }
      { uint2 p; p.x = pk_bf16(r0.y, r1.y); p.y = pk_bf16(r2.y, r3.y);
        *(uint2*)(&xT[(c0 + 1) * ST + j0]) = p; }
      { uint2 p; p.x = pk_bf16(r0.z, r1.z); p.y = pk_bf16(r2.z, r3.z);
        *(uint2*)(&xT[(c0 + 2) * ST + j0]) = p; }
      { uint2 p; p.x = pk_bf16(r0.w, r1.w); p.y = pk_bf16(r2.w, r3.w);
        *(uint2*)(&xT[(c0 + 3) * ST + j0]) = p; }
      atomicAdd(&sqs[j0 + 0], negsig * (r0.x*r0.x + r0.y*r0.y + r0.z*r0.z + r0.w*r0.w));
      atomicAdd(&sqs[j0 + 1], negsig * (r1.x*r1.x + r1.y*r1.y + r1.z*r1.z + r1.w*r1.w));
      atomicAdd(&sqs[j0 + 2], negsig * (r2.x*r2.x + r2.y*r2.y + r2.z*r2.z + r2.w*r2.w));
      atomicAdd(&sqs[j0 + 3], negsig * (r3.x*r3.x + r3.y*r3.y + r3.z*r3.z + r3.w*r3.w));
    }
    __syncthreads();

    // ---- GEMM1 (swapped): S[s'][m] tiles, Gaussian epilogue -> kt[wid][m][s'] ----
#pragma unroll
    for (int h = 0; h < 4; ++h) {
      const v16bf a_lo = load_frag_A(xs, h * 16 + ln, ST, 0,  half);
      const v16bf a_hi = load_frag_A(xs, h * 16 + ln, ST, 32, half);
      v8f sacc = {};
      sacc = __builtin_amdgcn_wmma_f32_16x16x32_bf16(false, a_lo, false, bm_lo,
                                                     (short)0, sacc, false, false);
      sacc = __builtin_amdgcn_wmma_f32_16x16x32_bf16(false, a_hi, false, bm_hi,
                                                     (short)0, sacc, false, false);
      // D layout: element r is s' = h*16 + half*8 + r, column m = ln
      const int sb = h * 16 + half * 8;
      const float4 bn0 = *(const float4*)(&sqs[sb]);
      const float4 bn1 = *(const float4*)(&sqs[sb + 4]);
      const float k0 = __expf(fminf(fmaf(sacc[0], twosig, am + bn0.x), 0.0f));
      const float k1 = __expf(fminf(fmaf(sacc[1], twosig, am + bn0.y), 0.0f));
      const float k2 = __expf(fminf(fmaf(sacc[2], twosig, am + bn0.z), 0.0f));
      const float k3 = __expf(fminf(fmaf(sacc[3], twosig, am + bn0.w), 0.0f));
      const float k4 = __expf(fminf(fmaf(sacc[4], twosig, am + bn1.x), 0.0f));
      const float k5 = __expf(fminf(fmaf(sacc[5], twosig, am + bn1.y), 0.0f));
      const float k6 = __expf(fminf(fmaf(sacc[6], twosig, am + bn1.z), 0.0f));
      const float k7 = __expf(fminf(fmaf(sacc[7], twosig, am + bn1.w), 0.0f));
      uint4 u;
      u.x = pk_bf16(k0, k1);
      u.y = pk_bf16(k2, k3);
      u.z = pk_bf16(k4, k5);
      u.w = pk_bf16(k6, k7);
      *(uint4*)(&kt[wid][ln * ST + sb]) = u;  // one b128 store per tile
    }

    // ---- GEMM2 (transposed out): accT[c][m] += xT(c,s') x K^T(s',m) ----
#pragma unroll
    for (int kb = 0; kb < 2; ++kb) {
      const v16bf bk = load_frag_B(kt[wid], ln, ST, kb * 32, half);
#pragma unroll
      for (int ct = 0; ct < 4; ++ct) {
        const v16bf a = load_frag_A(xT, ct * 16 + ln, ST, kb * 32, half);
        accT[ct] = __builtin_amdgcn_wmma_f32_16x16x32_bf16(false, a, false, bk,
                                                           (short)0, accT[ct], false, false);
      }
    }
  }

  // ---- epilogue: out[m][c] = x[m][c] + accT ; lane owns row m, r = consecutive c ----
  {
    const int m = m0 + wid * 16 + ln;
    const float* xr = x + (size_t)b * T * C + (size_t)m * C;
    float* orow = out + (size_t)b * T * C + (size_t)m * C;
#pragma unroll
    for (int ct = 0; ct < 4; ++ct) {
      const int cb = ct * 16 + half * 8;  // accT[ct][r] -> c = cb + r
      const float4 x0 = *(const float4*)(xr + cb);
      const float4 x1 = *(const float4*)(xr + cb + 4);
      float4 o0, o1;
      o0.x = x0.x + accT[ct][0]; o0.y = x0.y + accT[ct][1];
      o0.z = x0.z + accT[ct][2]; o0.w = x0.w + accT[ct][3];
      o1.x = x1.x + accT[ct][4]; o1.y = x1.y + accT[ct][5];
      o1.z = x1.z + accT[ct][6]; o1.w = x1.w + accT[ct][7];
      *(float4*)(orow + cb)     = o0;
      *(float4*)(orow + cb + 4) = o1;
    }
  }
}

extern "C" void kernel_launch(void* const* d_in, const int* in_sizes, int n_in,
                              void* d_out, int out_size, void* d_ws, size_t ws_size,
                              hipStream_t stream) {
  const float* x  = (const float*)d_in[0];
  const float* rs = (const float*)d_in[1];
  float* out = (float*)d_out;

  constexpr int T = 2048, C = 64, BM = 128;
  const int B = in_sizes[0] / (T * C);  // 32

  dim3 grid(T / BM, B);
  dim3 block(256);
  gauss_kernel_attn<<<grid, block, 0, stream>>>(x, rs, out);
}
